// Block_85126251807269
// MI455X (gfx1250) — matrix operand order repaired and checked by reading, not compiled, not run on hardware
//
#include <hip/hip_runtime.h>
#include <hip/hip_bf16.h>
#include <math.h>

// ---------------------------------------------------------------------------
// RetNet block for MI455X (gfx1250): bf16 WMMA pipeline, K-contiguous operand
// layouts so every fragment load is b128, double-buffered GEMM mainloop.
// ---------------------------------------------------------------------------

typedef __attribute__((ext_vector_type(16))) __bf16 v16bf;
typedef __attribute__((ext_vector_type(8)))  float  v8f;
typedef __attribute__((ext_vector_type(8)))  __bf16 bf16x8;

#define DEV __device__ __forceinline__

static constexpr int Bb   = 2;
static constexpr int Lb   = 4096;
static constexpr int Db   = 1024;
static constexpr int DKb  = 128;
static constexpr int DVb  = 256;
static constexpr int FFNb = 4096;
static constexpr int CH   = 64;       // chunk
static constexpr int NCH  = Lb / CH;  // 64 chunks
static constexpr int Mrows = Bb * Lb; // 8192
static constexpr int Hb   = 8;

DEV v8f v8f_zero() {
  v8f z;
#pragma unroll
  for (int i = 0; i < 8; ++i) z[i] = 0.f;
  return z;
}

DEV v8f wmma_bf16(v16bf a, v16bf b, v8f c) {
  return __builtin_amdgcn_wmma_f32_16x16x32_bf16(false, a, false, b,
                                                 (short)0, c, false, false);
}

// A fragment: logical A[16][32] from row-major storage (ld elements).
// lane<16: rows 0..15 K{0..7,16..23}; lane>=16: same rows K{8..15,24..31}
DEV v16bf load_a(const __bf16* base, int ld) {
  const int lane = threadIdx.x & 31;
  const __bf16* p = base + (lane & 15) * ld + ((lane >> 4) * 8);
  v16bf a;
#pragma unroll
  for (int e = 0; e < 8; ++e)  a[e] = p[e];
#pragma unroll
  for (int e = 8; e < 16; ++e) a[e] = p[e + 8];
  return a;
}

// B fragment: logical B[32][16] stored K-contiguous (i.e. Bt[n][k], ld elems).
// lane<16: K{0..7,16..23} of col lane; lane>=16: K{8..15,24..31} of col lane-16.
DEV v16bf load_b_ct(const __bf16* base, int ld) {
  const int lane = threadIdx.x & 31;
  const __bf16* p = base + (size_t)(lane & 15) * ld + ((lane >> 4) * 8);
  v16bf v;
#pragma unroll
  for (int e = 0; e < 8; ++e)  v[e] = p[e];
#pragma unroll
  for (int e = 8; e < 16; ++e) v[e] = p[e + 8];
  return v;
}

// ---------------------------------------------------------------------------
// weights: f32 [K][N] -> bf16 transposed [N][K]  (LDS-tiled 32x32)
// ---------------------------------------------------------------------------
__global__ __launch_bounds__(256) void cvt_transpose_kernel(
    const float* __restrict__ in, __bf16* __restrict__ out, int K, int N) {
  __shared__ float tile[32][33];
  const int n0 = blockIdx.x * 32, k0 = blockIdx.y * 32;
  const int tx = threadIdx.x & 31, ty = threadIdx.x >> 5;
#pragma unroll
  for (int i = ty; i < 32; i += 8)
    tile[i][tx] = in[(size_t)(k0 + i) * N + n0 + tx];
  __syncthreads();
#pragma unroll
  for (int i = ty; i < 32; i += 8)
    out[(size_t)(n0 + i) * K + k0 + tx] = (__bf16)tile[tx][i];
}

// ---------------------------------------------------------------------------
// LayerNorm (row = 1024), f32 in, bf16 out
// ---------------------------------------------------------------------------
__global__ __launch_bounds__(256) void layernorm_kernel(
    const float* __restrict__ x, const float* __restrict__ w,
    const float* __restrict__ bias, __bf16* __restrict__ out, int Dn) {
  const int row = blockIdx.x;
  const float* xr = x + (size_t)row * Dn;
  float s = 0.f, ss = 0.f;
  for (int i = threadIdx.x; i < Dn; i += 256) {
    float v = xr[i];
    s += v; ss += v * v;
  }
#pragma unroll
  for (int off = 16; off; off >>= 1) {
    s  += __shfl_xor(s,  off, 32);
    ss += __shfl_xor(ss, off, 32);
  }
  __shared__ float rs[8], rss[8];
  const int wid = threadIdx.x >> 5, lane = threadIdx.x & 31;
  if (lane == 0) { rs[wid] = s; rss[wid] = ss; }
  __syncthreads();
  if (threadIdx.x == 0) {
    float a = 0.f, aa = 0.f;
#pragma unroll
    for (int i = 0; i < 8; ++i) { a += rs[i]; aa += rss[i]; }
    rs[0] = a; rss[0] = aa;
  }
  __syncthreads();
  const float mean = rs[0] / Dn;
  const float var  = rss[0] / Dn - mean * mean;
  const float inv  = rsqrtf(var + 1e-6f);
  __bf16* orow = out + (size_t)row * Dn;
  for (int i = threadIdx.x; i < Dn; i += 256)
    orow[i] = (__bf16)((xr[i] - mean) * inv * w[i] + bias[i]);
}

// ---------------------------------------------------------------------------
// bf16 GEMM: C[M,N] = alpha * A[M,K] @ B[K,N] (+bias)(gelu)(+res)
// A row-major, B given TRANSPOSED (Bt[N][K]). Block tile 128x128, BK=32,
// 8 waves (2x4), wave tile 64x32. Double-buffered LDS, reg-prefetch pipeline.
// ---------------------------------------------------------------------------
#define GF_BIAS   1
#define GF_GELU   2
#define GF_RES    4
#define GF_OUTF32 8

__global__ __launch_bounds__(256) void gemm_bf16_kernel(
    const __bf16* __restrict__ A, const __bf16* __restrict__ Bt,
    const float* __restrict__ bias, const float* __restrict__ res,
    void* __restrict__ Cout, int M, int N, int K, float alpha, int flags) {
  __shared__ __bf16 sA[2][128][40];
  __shared__ __bf16 sB[2][128][40];
  const int tid  = threadIdx.x;
  const int lane = tid & 31;
  const int w    = tid >> 5;
  const int wm   = w >> 2;  // 0..1
  const int wn   = w & 3;   // 0..3
  const int m0   = blockIdx.y * 128;
  const int n0   = blockIdx.x * 128;
  const int row  = tid >> 1, seg = tid & 1;  // 128 rows x 2 segs of 16

  const __bf16* aSrc = A  + (size_t)(m0 + row) * K + seg * 16;
  const __bf16* bSrc = Bt + (size_t)(n0 + row) * K + seg * 16;

  bf16x8 ra0 = *(const bf16x8*)(aSrc);
  bf16x8 ra1 = *(const bf16x8*)(aSrc + 8);
  bf16x8 rb0 = *(const bf16x8*)(bSrc);
  bf16x8 rb1 = *(const bf16x8*)(bSrc + 8);

  v8f acc[4][2];
#pragma unroll
  for (int i = 0; i < 4; ++i)
#pragma unroll
    for (int j = 0; j < 2; ++j) acc[i][j] = v8f_zero();

  int cur = 0;
  for (int k0 = 0; k0 < K; k0 += 32) {
    *(bf16x8*)&sA[cur][row][seg * 16]     = ra0;
    *(bf16x8*)&sA[cur][row][seg * 16 + 8] = ra1;
    *(bf16x8*)&sB[cur][row][seg * 16]     = rb0;
    *(bf16x8*)&sB[cur][row][seg * 16 + 8] = rb1;
    __syncthreads();
    if (k0 + 32 < K) {  // prefetch next tile into registers
      ra0 = *(const bf16x8*)(aSrc + k0 + 32);
      ra1 = *(const bf16x8*)(aSrc + k0 + 40);
      rb0 = *(const bf16x8*)(bSrc + k0 + 32);
      rb1 = *(const bf16x8*)(bSrc + k0 + 40);
    }
    v16bf af[4], bfv[2];
#pragma unroll
    for (int i = 0; i < 4; ++i) af[i] = load_a(&sA[cur][wm * 64 + i * 16][0], 40);
#pragma unroll
    for (int j = 0; j < 2; ++j) bfv[j] = load_b_ct(&sB[cur][wn * 32 + j * 16][0], 40);
#pragma unroll
    for (int i = 0; i < 4; ++i)
#pragma unroll
      for (int j = 0; j < 2; ++j)
        acc[i][j] = wmma_bf16(af[i], bfv[j], acc[i][j]);
    cur ^= 1;
  }

  const int hi = lane >> 4, cl = lane & 15;
#pragma unroll
  for (int i = 0; i < 4; ++i) {
#pragma unroll
    for (int j = 0; j < 2; ++j) {
      const int col = n0 + wn * 32 + j * 16 + cl;
#pragma unroll
      for (int r = 0; r < 8; ++r) {
        const int rowg = m0 + wm * 64 + i * 16 + hi * 8 + r;
        float v = acc[i][j][r] * alpha;
        if (flags & GF_BIAS) v += bias[col];
        if (flags & GF_GELU) v = 0.5f * v * (1.f + erff(v * 0.70710678118654752f));
        if (flags & GF_RES)  v += res[(size_t)rowg * N + col];
        if (flags & GF_OUTF32) ((float*)Cout)[(size_t)rowg * N + col] = v;
        else ((__bf16*)Cout)[(size_t)rowg * N + col] = (__bf16)v;
      }
    }
  }
}

// ---------------------------------------------------------------------------
// Retention phase 1: G_i[dv][dk] = sum_s v[s,dv] * (k*d_k)[s,dk]   (TRANSPOSED
// output layout [dv][dk] so the scan and phase 3 stay K-contiguous).
// one block per (bh, chunk); wave w owns dk rows [16w,16w+16)
// ---------------------------------------------------------------------------
__global__ __launch_bounds__(256) void chunk_kv_kernel(
    const __bf16* __restrict__ kbf, const __bf16* __restrict__ vbf,
    __bf16* __restrict__ Gbf) {
  const int blk = blockIdx.x;
  const int ci  = blk & (NCH - 1);
  const int bh  = blk >> 6;
  const int b = bh >> 3, h = bh & 7;
  __shared__ __bf16 sKdT[DKb][CH + 8];  // (k*d_k)^T : [dk][s]  18 KB
  __shared__ __bf16 sVT[DVb][CH + 8];   // v^T       : [dv][s]  36 KB
  const float bdec = 1.f - exp2f(-5.f - (float)h);
  const float l2b  = log2f(bdec);
  const int tid = threadIdx.x, lane = tid & 31, w = tid >> 5;

  for (int idx = tid; idx < CH * DKb; idx += 256) {
    const int s = idx >> 7, dk = idx & 127;
    const float kv =
        (float)kbf[(size_t)(b * Lb + ci * CH + s) * Db + h * DKb + dk];
    sKdT[dk][s] = (__bf16)(kv * exp2f(l2b * (float)(CH - 1 - s)));
  }
  for (int idx = tid; idx < CH * DVb; idx += 256) {
    const int s = idx >> 8, dv = idx & 255;
    sVT[dv][s] = vbf[(size_t)(b * Lb + ci * CH + s) * (2 * Db) + h * DVb + dv];
  }
  __syncthreads();

  v8f acc[16];
#pragma unroll
  for (int t = 0; t < 16; ++t) acc[t] = v8f_zero();
  for (int ks = 0; ks < CH; ks += 32) {
    v16bf a = load_a(&sKdT[16 * w][ks], CH + 8);  // A[dk][s]
#pragma unroll
    for (int nt = 0; nt < 16; ++nt) {
      v16bf bv = load_b_ct(&sVT[nt * 16][ks], CH + 8);  // B[s][dv] K-contig
      acc[nt] = wmma_bf16(a, bv, acc[nt]);
    }
  }
  const int hi = lane >> 4, cl = lane & 15;
  const size_t gbase = ((size_t)bh * NCH + ci) * (DKb * DVb);
#pragma unroll
  for (int nt = 0; nt < 16; ++nt) {
    const int dv = nt * 16 + cl;
#pragma unroll
    for (int r = 0; r < 8; ++r) {
      const int dk = 16 * w + hi * 8 + r;
      Gbf[gbase + (size_t)dv * DKb + dk] = (__bf16)acc[nt][r];  // [dv][dk]
    }
  }
}

// ---------------------------------------------------------------------------
// Retention phase 2: exclusive decayed prefix scan over chunks (layout [dv][dk])
// ---------------------------------------------------------------------------
__global__ __launch_bounds__(256) void state_scan_kernel(
    const __bf16* __restrict__ Gbf, __bf16* __restrict__ Sbf) {
  const int bh = blockIdx.y;
  const int p  = blockIdx.x * 256 + threadIdx.x;  // 0..32767 (dv*128+dk)
  const int h  = bh & 7;
  const float bdec = 1.f - exp2f(-5.f - (float)h);
  const float dc   = exp2f(log2f(bdec) * (float)CH);
  const size_t base = (size_t)bh * NCH * (DKb * DVb) + p;
  float S = 0.f;
  for (int i = 0; i < NCH; ++i) {
    const size_t off = base + (size_t)i * (DKb * DVb);
    Sbf[off] = (__bf16)S;
    S = S * dc + (float)Gbf[off];
  }
}

// ---------------------------------------------------------------------------
// Retention phase 3: per-chunk output
//   A = (q k^T) ∘ mask ;  o = diag(d_q)·(q @ S_prev) + A @ v
// one block per (bh, chunk); wave w owns dv cols [32w, 32w+32)
// S stored [dv][dk] (K-contiguous); v staged transposed in LDS.
// ---------------------------------------------------------------------------
__global__ __launch_bounds__(256) void chunk_out_kernel(
    const __bf16* __restrict__ qbf, const __bf16* __restrict__ kbf,
    const __bf16* __restrict__ vbf, const __bf16* __restrict__ Sbf,
    __bf16* __restrict__ obf) {
  const int blk = blockIdx.x;
  const int ci  = blk & (NCH - 1);
  const int bh  = blk >> 6;
  const int b = bh >> 3, h = bh & 7;
  __shared__ __bf16 sQ[CH][DKb + 8];   // q rows (A operand)       17.4 KB
  __shared__ __bf16 sVT[DVb][CH + 8];  // v^T (B operand K-contig) 36.9 KB
  __shared__ __bf16 sA[CH][CH + 8];    // masked scores (A operand) 9.2 KB
  const float bdec = 1.f - exp2f(-5.f - (float)h);
  const float l2b  = log2f(bdec);
  const int tid = threadIdx.x, lane = tid & 31, w = tid >> 5;
  const int hi = lane >> 4, cl = lane & 15;

  const size_t rowbase = (size_t)(b * Lb + ci * CH);
  for (int idx = tid; idx < CH * DKb; idx += 256) {
    const int s = idx >> 7, dk = idx & 127;
    sQ[s][dk] = qbf[(rowbase + s) * Db + h * DKb + dk];
  }
  for (int idx = tid; idx < CH * DVb; idx += 256) {
    const int s = idx >> 8, dv = idx & 255;
    sVT[dv][s] = vbf[(rowbase + s) * (2 * Db) + h * DVb + dv];
  }
  __syncthreads();

  // ---- intra-chunk scores A (16 tiles of 16x16, 2 per wave) ----
  const __bf16* kg = kbf + rowbase * Db + h * DKb;  // k[s][dk]: K-contig for B
#pragma unroll
  for (int tt = 0; tt < 2; ++tt) {
    const int t = w * 2 + tt, mt = t >> 2, nt = t & 3;
    v8f a = v8f_zero();
#pragma unroll
    for (int ks = 0; ks < DKb; ks += 32) {
      v16bf af = load_a(&sQ[mt * 16][ks], DKb + 8);
      v16bf bv = load_b_ct(kg + (size_t)(nt * 16) * Db + ks, Db);  // k^T
      a = wmma_bf16(af, bv, a);
    }
    const int sg = nt * 16 + cl;
#pragma unroll
    for (int r = 0; r < 8; ++r) {
      const int tg = mt * 16 + hi * 8 + r;
      const float d = (float)(tg - sg);
      sA[tg][sg] = (__bf16)((tg >= sg) ? a[r] * exp2f(l2b * d) : 0.f);
    }
  }
  __syncthreads();

  // ---- cross-chunk: q @ S_prev, then row-scale by d_q = b^(t+1) ----
  v8f o[4][2];
#pragma unroll
  for (int i = 0; i < 4; ++i)
#pragma unroll
    for (int j = 0; j < 2; ++j) o[i][j] = v8f_zero();

  const size_t sbase = ((size_t)bh * NCH + ci) * (DKb * DVb);
#pragma unroll
  for (int mt = 0; mt < 4; ++mt) {
#pragma unroll
    for (int ks = 0; ks < DKb; ks += 32) {
      v16bf af = load_a(&sQ[mt * 16][ks], DKb + 8);
#pragma unroll
      for (int j = 0; j < 2; ++j) {
        v16bf bv = load_b_ct(
            Sbf + sbase + (size_t)(w * 32 + j * 16) * DKb + ks, DKb);
        o[mt][j] = wmma_bf16(af, bv, o[mt][j]);
      }
    }
  }
  // row scale (d_q factors out of the K-sum)
#pragma unroll
  for (int mt = 0; mt < 4; ++mt) {
#pragma unroll
    for (int r = 0; r < 8; ++r) {
      const float dq = exp2f(l2b * (float)(mt * 16 + hi * 8 + r + 1));
#pragma unroll
      for (int j = 0; j < 2; ++j) o[mt][j][r] *= dq;
    }
  }
  // ---- intra-chunk: += A @ v ----
#pragma unroll
  for (int mt = 0; mt < 4; ++mt) {
#pragma unroll
    for (int ks = 0; ks < CH; ks += 32) {
      v16bf af = load_a(&sA[mt * 16][ks], CH + 8);
#pragma unroll
      for (int j = 0; j < 2; ++j) {
        v16bf bv = load_b_ct(&sVT[w * 32 + j * 16][ks], CH + 8);
        o[mt][j] = wmma_bf16(af, bv, o[mt][j]);
      }
    }
  }
  // store o (bf16) into [B, L, H*DV]
#pragma unroll
  for (int mt = 0; mt < 4; ++mt)
#pragma unroll
    for (int j = 0; j < 2; ++j)
#pragma unroll
      for (int r = 0; r < 8; ++r) {
        const int tg = mt * 16 + hi * 8 + r;
        const int dv = w * 32 + j * 16 + cl;
        obf[(rowbase + tg) * (2 * Db) + h * DVb + dv] = (__bf16)o[mt][j][r];
      }
}

// ---------------------------------------------------------------------------
// Orchestration
// ---------------------------------------------------------------------------
extern "C" void kernel_launch(void* const* d_in, const int* in_sizes, int n_in,
                              void* d_out, int out_size, void* d_ws,
                              size_t ws_size, hipStream_t stream) {
  (void)in_sizes; (void)n_in; (void)out_size; (void)ws_size;
  const float* x    = (const float*)d_in[0];
  const float* ln1w = (const float*)d_in[1];
  const float* ln1b = (const float*)d_in[2];
  const float* Wq   = (const float*)d_in[3];
  const float* Wk   = (const float*)d_in[4];
  const float* Wv   = (const float*)d_in[5];
  const float* Wo   = (const float*)d_in[6];
  const float* ln2w = (const float*)d_in[7];
  const float* ln2b = (const float*)d_in[8];
  const float* W1   = (const float*)d_in[9];
  const float* b1   = (const float*)d_in[10];
  const float* W2   = (const float*)d_in[11];
  const float* b2   = (const float*)d_in[12];

  char* ws = (char*)d_ws;
  const size_t MB = 1ull << 20;
  // workspace layout (bytes); all weight buffers hold W^T (bf16, [N][K])
  __bf16* WqT = (__bf16*)(ws + 0 * MB);    //  2 MB
  __bf16* WkT = (__bf16*)(ws + 2 * MB);    //  2 MB
  __bf16* WvT = (__bf16*)(ws + 4 * MB);    //  4 MB
  __bf16* WoT = (__bf16*)(ws + 8 * MB);    //  4 MB
  __bf16* W1T = (__bf16*)(ws + 12 * MB);   //  8 MB
  __bf16* W2T = (__bf16*)(ws + 20 * MB);   //  8 MB
  __bf16* hB  = (__bf16*)(ws + 28 * MB);   // 16 MB (ln1 out; reused for ln2)
  __bf16* qB  = (__bf16*)(ws + 44 * MB);   // 16 MB
  __bf16* kB  = (__bf16*)(ws + 60 * MB);   // 16 MB
  __bf16* vB  = (__bf16*)(ws + 76 * MB);   // 32 MB
  __bf16* GB  = (__bf16*)(ws + 108 * MB);  // 64 MB (reused as FFN act)
  __bf16* SB  = (__bf16*)(ws + 172 * MB);  // 64 MB
  __bf16* oB  = (__bf16*)(ws + 236 * MB);  // 32 MB
  float*  x1  = (float*)(ws + 44 * MB);    // 32 MB f32, aliases qB+kB (dead)
  __bf16* fB  = GB;                        // FFN activation, aliases GB (dead)

  // 1) weights -> bf16 transposed
  cvt_transpose_kernel<<<dim3(Db / 32, Db / 32), 256, 0, stream>>>(Wq, WqT, Db, Db);
  cvt_transpose_kernel<<<dim3(Db / 32, Db / 32), 256, 0, stream>>>(Wk, WkT, Db, Db);
  cvt_transpose_kernel<<<dim3((2 * Db) / 32, Db / 32), 256, 0, stream>>>(Wv, WvT, Db, 2 * Db);
  cvt_transpose_kernel<<<dim3(Db / 32, (2 * Db) / 32), 256, 0, stream>>>(Wo, WoT, 2 * Db, Db);
  cvt_transpose_kernel<<<dim3(FFNb / 32, Db / 32), 256, 0, stream>>>(W1, W1T, Db, FFNb);
  cvt_transpose_kernel<<<dim3(Db / 32, FFNb / 32), 256, 0, stream>>>(W2, W2T, FFNb, Db);

  // 2) LN1
  layernorm_kernel<<<Mrows, 256, 0, stream>>>(x, ln1w, ln1b, hB, Db);

  // 3) q/k/v projections (q folds DK^-0.5)
  gemm_bf16_kernel<<<dim3(Db / 128, Mrows / 128), 256, 0, stream>>>(
      hB, WqT, nullptr, nullptr, qB, Mrows, Db, Db, 0.08838834764831845f, 0);
  gemm_bf16_kernel<<<dim3(Db / 128, Mrows / 128), 256, 0, stream>>>(
      hB, WkT, nullptr, nullptr, kB, Mrows, Db, Db, 1.f, 0);
  gemm_bf16_kernel<<<dim3((2 * Db) / 128, Mrows / 128), 256, 0, stream>>>(
      hB, WvT, nullptr, nullptr, vB, Mrows, 2 * Db, Db, 1.f, 0);

  // 4) retention: chunk KV -> decayed scan -> chunk out
  chunk_kv_kernel<<<Bb * Hb * NCH, 256, 0, stream>>>(kB, vB, GB);
  state_scan_kernel<<<dim3((DKb * DVb) / 256, Bb * Hb), 256, 0, stream>>>(GB, SB);
  chunk_out_kernel<<<Bb * Hb * NCH, 256, 0, stream>>>(qB, kB, vB, SB, oB);

  // 5) x1 = x + o @ Wo   (f32 out, overwrites dead q/k region)
  gemm_bf16_kernel<<<dim3(Db / 128, Mrows / 128), 256, 0, stream>>>(
      oB, WoT, nullptr, x, x1, Mrows, Db, 2 * Db, 1.f, GF_RES | GF_OUTF32);

  // 6) LN2 (reuse hB)
  layernorm_kernel<<<Mrows, 256, 0, stream>>>(x1, ln2w, ln2b, hB, Db);

  // 7) FFN: f = gelu(h @ W1 + b1) (bf16, reuse GB)
  gemm_bf16_kernel<<<dim3(FFNb / 128, Mrows / 128), 256, 0, stream>>>(
      hB, W1T, b1, nullptr, fB, Mrows, FFNb, Db, 1.f, GF_BIAS | GF_GELU);

  // 8) out = x1 + f @ W2 + b2 (f32 -> d_out)
  gemm_bf16_kernel<<<dim3(Db / 128, Mrows / 128), 256, 0, stream>>>(
      fB, W2T, b2, x1, d_out, Mrows, Db, FFNb, 1.f,
      GF_BIAS | GF_RES | GF_OUTF32);
}
